// MultiHeadAttention_1202590843545
// MI455X (gfx1250) — matrix operand order repaired and checked by reading, not compiled, not run on hardware
//
#include <hip/hip_runtime.h>

typedef __attribute__((ext_vector_type(16))) _Float16 v16h;
typedef __attribute__((ext_vector_type(8)))  float    v8f;

#define B_   2
#define T_   2048
#define D_   1024
#define H_   16
#define DH_  64
#define SCALE_ 0.125f   // DH^-0.5

// ---------------- CDNA5 wave32 WMMA fragment loaders (16-bit operands) ----------------
// A-matrix 16x32: lane holds row m = lane&15; VGPR halves hold K with the
// documented interleave (K 0..7 / 16..23 for lanes 0-15, +8 for lanes 16-31).
__device__ __forceinline__ v16h ld_frag_a(const _Float16* p, int ld, int lane) {
  const int m = lane & 15, hi = lane >> 4;
  const _Float16* row = p + m * ld;
  v16h a;
#pragma unroll
  for (int v = 0; v < 8; ++v) {
    const int k = (v < 4) ? (hi * 8 + 2 * v) : (16 + hi * 8 + 2 * (v - 4));
    a[2 * v]     = row[k];
    a[2 * v + 1] = row[k + 1];
  }
  return a;
}

// B-matrix 32x16, LDS tile stored N-major (row = n, K contiguous): lane = n,
// same K interleave as A (K 0..7 / 16..23 for lanes 0-15, +8 for lanes 16-31).
__device__ __forceinline__ v16h ld_frag_b_nmajor(const _Float16* p, int ld, int lane) {
  const int n = lane & 15, hi = lane >> 4;
  const _Float16* row = p + n * ld;
  v16h b;
#pragma unroll
  for (int v = 0; v < 8; ++v) {
    const int k = (v < 4) ? (hi * 8 + 2 * v) : (16 + hi * 8 + 2 * (v - 4));
    b[2 * v]     = row[k];
    b[2 * v + 1] = row[k + 1];
  }
  return b;
}

__device__ __forceinline__ v8f wmma_f16(v16h a, v16h b, v8f c) {
  return __builtin_amdgcn_wmma_f32_16x16x32_f16(false, a, false, b, (short)0, c,
                                                false, false);
}

// ---------------- GEMM: C[M,N] = A[M,K] * W[N,K]^T  (y = x @ W.T) ----------------
// M=4096 (B*T), N=K=1024.  Block tile 128x64, BK=64, 8 waves, wave tile 32x32.
template <typename TA, typename TC>
__global__ __launch_bounds__(256) void gemm_nt(const TA* __restrict__ A,
                                               const float* __restrict__ W,
                                               TC* __restrict__ C) {
  constexpr int N = D_, K = D_;
  constexpr int BM = 128, BN = 64, BK = 64;
  constexpr int LDA = BK + 8, LDB = BK + 8;          // 144B row stride (16B aligned)
  __shared__ _Float16 As[BM * LDA];                  // 18 KB
  __shared__ _Float16 Bs[BN * LDB];                  //  9 KB
  const int tid = threadIdx.x, lane = tid & 31, w = tid >> 5;
  const int wm = w >> 1, wn = w & 1;
  const int bm = blockIdx.y * BM, bn = blockIdx.x * BN;
  v8f c[2][2] = {};
  const int r0 = tid >> 4;          // 0..15
  const int c0 = (tid & 15) * 4;    // 0,4,...,60

  for (int kb = 0; kb < K; kb += BK) {
    __syncthreads();
#pragma unroll
    for (int i = 0; i < 8; ++i) {   // A tile: 128x64 -> f16
      const int r = r0 + i * 16;
      const TA* src = A + (size_t)(bm + r) * K + kb + c0;
      _Float16* dst = As + r * LDA + c0;
#pragma unroll
      for (int e = 0; e < 4; ++e) dst[e] = (_Float16)src[e];
    }
#pragma unroll
    for (int i = 0; i < 4; ++i) {   // W tile: 64x64, f32 -> f16
      const int r = r0 + i * 16;
      const float* src = W + (size_t)(bn + r) * K + kb + c0;
      _Float16* dst = Bs + r * LDB + c0;
#pragma unroll
      for (int e = 0; e < 4; ++e) dst[e] = (_Float16)src[e];
    }
    __syncthreads();
    const _Float16* Ab = As + (wm * 32) * LDA;
    const _Float16* Bb = Bs + (wn * 32) * LDB;
#pragma unroll
    for (int s = 0; s < 2; ++s) {   // two K=32 WMMA steps per stage
      v16h a0 = ld_frag_a(Ab + s * 32, LDA, lane);
      v16h a1 = ld_frag_a(Ab + 16 * LDA + s * 32, LDA, lane);
      v16h b0 = ld_frag_b_nmajor(Bb + s * 32, LDB, lane);
      v16h b1 = ld_frag_b_nmajor(Bb + 16 * LDB + s * 32, LDB, lane);
      c[0][0] = wmma_f16(a0, b0, c[0][0]);
      c[0][1] = wmma_f16(a0, b1, c[0][1]);
      c[1][0] = wmma_f16(a1, b0, c[1][0]);
      c[1][1] = wmma_f16(a1, b1, c[1][1]);
    }
  }

  const int ln = lane & 15, hi = lane >> 4;
#pragma unroll
  for (int i = 0; i < 2; ++i)
#pragma unroll
    for (int j = 0; j < 2; ++j)
#pragma unroll
      for (int v = 0; v < 8; ++v) {
        const int m = bm + wm * 32 + i * 16 + v + hi * 8;
        const int n = bn + wn * 32 + j * 16 + ln;
        C[(size_t)m * N + n] = (TC)c[i][j][v];
      }
}

// ---------------- Flash attention (causal + key padding), f16 WMMA, f32 accum ----------------
// Grid: (T/64, B*H). Block: 128 threads = 4 waves; wave w owns q-rows [w*16, w*16+16).
__global__ __launch_bounds__(128) void flash_attn(const _Float16* __restrict__ Qh,
                                                  const _Float16* __restrict__ Kh,
                                                  const _Float16* __restrict__ Vh,
                                                  const unsigned char* __restrict__ kpad,
                                                  _Float16* __restrict__ Ctx) {
  constexpr int QT = 64, KT = 64, LDH = DH_ + 8, LDV = KT + 8, LDP = KT + 8;
  __shared__ _Float16 Qs[QT * LDH];
  __shared__ _Float16 Ks[KT * LDH];        // [t][d]
  __shared__ _Float16 Vt[DH_ * LDV];       // [d][t]  (transposed at fill)
  __shared__ _Float16 Ps[4 * 16 * LDP];    // per-wave P tiles
  const int tid = threadIdx.x, lane = tid & 31, w = tid >> 5;
  const int qt = blockIdx.x;
  const int bb = blockIdx.y / H_, h = blockIdx.y % H_;
  const int ln = lane & 15, hi = lane >> 4;

  { // load Q tile 64x64
    const int r = tid >> 1, cc = (tid & 1) * 32;
    const _Float16* src = Qh + ((size_t)bb * T_ + qt * QT + r) * D_ + h * DH_ + cc;
    _Float16* dst = Qs + r * LDH + cc;
#pragma unroll
    for (int e = 0; e < 32; ++e) dst[e] = src[e];
  }
  __syncthreads();
  const v16h qf0 = ld_frag_a(Qs + (w * 16) * LDH, LDH, lane);
  const v16h qf1 = ld_frag_a(Qs + (w * 16) * LDH + 32, LDH, lane);

  float mi[8], li[8];
  v8f co[4] = {};
#pragma unroll
  for (int v = 0; v < 8; ++v) { mi[v] = -1e30f; li[v] = 0.0f; }

  const int qrow0 = qt * QT + w * 16;

  auto do_tile = [&](int kt, bool causal) {
    __syncthreads();
    { // load K row-major, V transposed (d-major) into LDS
      const int r = tid >> 1, cc = (tid & 1) * 32;
      const size_t base = ((size_t)bb * T_ + kt * KT + r) * D_ + h * DH_ + cc;
      const _Float16* sk = Kh + base;
      const _Float16* svv = Vh + base;
      _Float16* dk = Ks + r * LDH + cc;
#pragma unroll
      for (int e = 0; e < 32; ++e) dk[e] = sk[e];
#pragma unroll
      for (int e = 0; e < 32; ++e) Vt[(cc + e) * LDV + r] = svv[e];
      if (kt < qt) {                       // prefetch next tile (global_prefetch_b8)
        __builtin_prefetch(Kh + base + (size_t)KT * D_, 0, 1);
        __builtin_prefetch(Vh + base + (size_t)KT * D_, 0, 1);
      }
    }
    __syncthreads();

    // S = Q * K^T  (K-dim = DH = 64 -> 2 WMMA steps, 4 key sub-tiles)
    v8f cs[4] = {};
#pragma unroll
    for (int j = 0; j < 4; ++j) {
      v16h bk0 = ld_frag_b_nmajor(Ks + (j * 16) * LDH, LDH, lane);
      v16h bk1 = ld_frag_b_nmajor(Ks + (j * 16) * LDH + 32, LDH, lane);
      cs[j] = wmma_f16(qf0, bk0, cs[j]);
      cs[j] = wmma_f16(qf1, bk1, cs[j]);
    }

    // key-padding as a per-column additive bias; causal compare only on diag tile
    float padf[4];
#pragma unroll
    for (int j = 0; j < 4; ++j)
      padf[j] = (kpad[(size_t)bb * T_ + kt * KT + j * 16 + ln] != 0) ? -1e30f : 0.0f;

    float sv[4][8], rowmax[8];
#pragma unroll
    for (int v = 0; v < 8; ++v) rowmax[v] = -1e30f;
#pragma unroll
    for (int j = 0; j < 4; ++j) {
#pragma unroll
      for (int v = 0; v < 8; ++v) {
        float x = cs[j][v] * SCALE_ + padf[j];
        if (causal) {
          const int kg = kt * KT + j * 16 + ln;
          const int qg = qrow0 + v + hi * 8;
          if (kg > qg) x = -1e30f;
        }
        sv[j][v] = x;
        rowmax[v] = fmaxf(rowmax[v], x);
      }
    }
#pragma unroll
    for (int v = 0; v < 8; ++v)
#pragma unroll
      for (int m = 1; m < 16; m <<= 1)
        rowmax[v] = fmaxf(rowmax[v], __shfl_xor(rowmax[v], m, 32));

    float alpha[8], rs[8];
#pragma unroll
    for (int v = 0; v < 8; ++v) {
      const float mn = fmaxf(mi[v], rowmax[v]);
      alpha[v] = __expf(mi[v] - mn);
      mi[v] = mn;
      rs[v] = 0.0f;
    }
#pragma unroll
    for (int j = 0; j < 4; ++j)
#pragma unroll
      for (int v = 0; v < 8; ++v) {
        const float p = __expf(sv[j][v] - mi[v]);
        sv[j][v] = p;
        rs[v] += p;
      }
#pragma unroll
    for (int v = 0; v < 8; ++v) {
#pragma unroll
      for (int m = 1; m < 16; m <<= 1) rs[v] += __shfl_xor(rs[v], m, 32);
      li[v] = li[v] * alpha[v] + rs[v];
    }
#pragma unroll
    for (int j = 0; j < 4; ++j)
#pragma unroll
      for (int v = 0; v < 8; ++v) co[j][v] *= alpha[v];

    // stash P into this wave's LDS region (LDS is in-order per wave)
    _Float16* pw = Ps + (w * 16) * LDP;
#pragma unroll
    for (int j = 0; j < 4; ++j)
#pragma unroll
      for (int v = 0; v < 8; ++v)
        pw[(v + hi * 8) * LDP + j * 16 + ln] = (_Float16)sv[j][v];

    // O += P * V  (K-dim = 64 keys -> 2 WMMA steps; V^T tile -> contiguous frags)
#pragma unroll
    for (int s = 0; s < 2; ++s) {
      v16h ap = ld_frag_a(pw + s * 32, LDP, lane);
#pragma unroll
      for (int j = 0; j < 4; ++j) {
        v16h bv = ld_frag_b_nmajor(Vt + (j * 16) * LDV + s * 32, LDV, lane);
        co[j] = wmma_f16(ap, bv, co[j]);
      }
    }
  };

  for (int kt = 0; kt < qt; ++kt) do_tile(kt, false);  // full tiles: no causal compare
  do_tile(qt, true);                                   // diagonal tile

  // normalize and store context (f16) for the Wo projection
#pragma unroll
  for (int v = 0; v < 8; ++v) {
    const float inv = 1.0f / li[v];
    const int qg = qrow0 + v + hi * 8;
    _Float16* dst = Ctx + ((size_t)bb * T_ + qg) * D_ + h * DH_;
#pragma unroll
    for (int j = 0; j < 4; ++j) dst[j * 16 + ln] = (_Float16)(co[j][v] * inv);
  }
}

// ---------------- host launcher ----------------
extern "C" void kernel_launch(void* const* d_in, const int* in_sizes, int n_in,
                              void* d_out, int out_size, void* d_ws, size_t ws_size,
                              hipStream_t stream) {
  (void)in_sizes; (void)n_in; (void)out_size; (void)ws_size;
  const float* query = (const float*)d_in[0];
  const float* key   = (const float*)d_in[1];
  const float* value = (const float*)d_in[2];
  // d_in[3] = [T,T] causal mask: computed analytically in-kernel, not read.
  const unsigned char* kpad = (const unsigned char*)d_in[4];   // [B,T] bool
  const float* Wq = (const float*)d_in[5];
  const float* Wk = (const float*)d_in[6];
  const float* Wv = (const float*)d_in[7];
  const float* Wo = (const float*)d_in[8];
  float* out = (float*)d_out;

  const size_t MT = (size_t)B_ * T_ * D_;           // 4 Mi elements
  _Float16* Qh  = (_Float16*)d_ws;                  //  8 MB each
  _Float16* Kh  = Qh + MT;
  _Float16* Vh  = Kh + MT;
  _Float16* Ctx = Vh + MT;                          // total 32 MB workspace

  dim3 gg(D_ / 64, (B_ * T_) / 128);                // (16, 32)
  gemm_nt<float, _Float16><<<gg, 256, 0, stream>>>(query, Wq, Qh);
  gemm_nt<float, _Float16><<<gg, 256, 0, stream>>>(key,   Wk, Kh);
  gemm_nt<float, _Float16><<<gg, 256, 0, stream>>>(value, Wv, Vh);

  dim3 gf(T_ / 64, B_ * H_);                        // (32, 32)
  flash_attn<<<gf, 128, 0, stream>>>(Qh, Kh, Vh, kpad, Ctx);

  gemm_nt<_Float16, float><<<gg, 256, 0, stream>>>(Ctx, Wo, out);
}
